// RUOT_29308856828411
// MI455X (gfx1250) — matrix operand order repaired and checked by reading, not compiled, not run on hardware
//
#include <hip/hip_runtime.h>
#include <hip/hip_bf16.h>
#include <stdint.h>

// ---------------------------------------------------------------------------
// CNF exact-trace step on MI455X (gfx1250, wave32).
//
// All GEMMs route through v_wmma_f32_16x16x32_bf16 (bf16 in, f32 accumulate).
// Roofline: ~275 GFLOP of the ~302 GFLOP total is the exact Jacobian trace
// (per-sample [32x512]@[512x512] twice). Weights (~3.4 MB bf16) are L2
// resident (192 MB L2). Design:
//   * M=128 row blocks, 8 waves/block covering 128 output cols, K-paneled
//     LDS staging (128x64 bf16 = 16 KB) -> weight reads amortized 8 M-tiles
//     deep, A panels amortized across 8 waves.
//   * Inner loop: batch all 8 LDS A-fragments into registers, then issue 8
//     WMMAs against one streamed B-fragment (avoids ds_load->wmma stalls).
//   * Jacobian trace as chunked global batched GEMMs (1024 samples/chunk,
//     2x32 MB ping-pong in d_ws) with a sech^2-scaling epilogue.
//
// d_in: 0:t 1:z | params1: 2:W1 3:b1 4:W2 5:b2 6:W3 7:b3 8:W4 9:b4
//       | params2: 10..17 | params3: 18..25
// d_out: dz_dt[B*32] | g[B] | dlogp[B] | dlog_p_dz[B*32]
// Workspace: ~110 MB.
// ---------------------------------------------------------------------------

#define B_N   8192
#define D_N   32
#define H_N   512
#define CHUNK 1024          // samples per tangent chunk
#define NCHUNK (B_N / CHUNK)

typedef __bf16 bf16_t;
typedef bf16_t v16bf __attribute__((ext_vector_type(16)));
typedef float  v8f   __attribute__((ext_vector_type(8)));

__device__ __forceinline__ float bf2f(unsigned short u) {
  union { unsigned int i; float f; } c; c.i = ((unsigned int)u) << 16; return c.f;
}
__device__ __forceinline__ unsigned short f2bf(float f) {
  union { float f; unsigned int i; } c; c.f = f;
  unsigned int r = c.i + 0x7FFFu + ((c.i >> 16) & 1u);
  return (unsigned short)(r >> 16);
}

union Frag { v16bf v; uint4 q[2]; };
union Pk8  { uint4 q; unsigned short s[8]; };

// A fragment (16 rows x 32 K), row-major source with given element stride.
// Lane m<16: row m, K = {k0+0..7, k0+16..23}; lanes 16-31: +8 on K.
__device__ __forceinline__ v16bf load_a_frag(const unsigned short* base, int stride,
                                             int k0, int lane) {
  const int r = lane & 15, hi = lane >> 4;
  const unsigned short* p = base + (size_t)r * stride + k0 + 8 * hi;
  Frag f;
  f.q[0] = *(const uint4*)(p);
  f.q[1] = *(const uint4*)(p + 16);
  return f.v;
}

// B fragment from W^T [N, KPAD] row-major (column n of W = row n of W^T).
// Lane n<16: col n, K = {k0+0..7, k0+16..23}; lanes 16-31: +8 on K.
__device__ __forceinline__ v16bf load_b_frag(const unsigned short* wt, int kpad,
                                             int ncol0, int k0, int lane) {
  const int n = lane & 15, hi = lane >> 4;
  const unsigned short* p = wt + (size_t)(ncol0 + n) * kpad + k0 + 8 * hi;
  Frag f;
  f.q[0] = *(const uint4*)(p);
  f.q[1] = *(const uint4*)(p + 16);
  return f.v;
}

// ------------------------- prep kernels ------------------------------------

__global__ void k_build_state(const float* t, const float* z, unsigned short* st) {
  int idx = blockIdx.x * 256 + threadIdx.x;
  if (idx >= B_N * 64) return;
  int b = idx >> 6, c = idx & 63;
  float v = (c == 0) ? t[0] : ((c <= D_N) ? z[(size_t)b * D_N + (c - 1)] : 0.0f);
  st[idx] = f2bf(v);
}

// WT[n][k] = W[k][n], zero-padded to [NPAD, KPAD], bf16.
__global__ void k_transpose(const float* W, unsigned short* WT,
                            int K, int Nsrc, int KPAD, int NPAD) {
  int idx = blockIdx.x * 256 + threadIdx.x;
  if (idx >= NPAD * KPAD) return;
  int n = idx / KPAD, k = idx - n * KPAD;
  float v = (n < Nsrc && k < K) ? W[(size_t)k * Nsrc + n] : 0.0f;
  WT[idx] = f2bf(v);
}

// W1z (rows 1..32 of W1 [33,512]) cast to bf16, row-major [32,512].
__global__ void k_cast_w1z(const float* W1, unsigned short* W1zb) {
  int idx = blockIdx.x * 256 + threadIdx.x;
  if (idx >= 32 * H_N) return;
  W1zb[idx] = f2bf(W1[H_N + idx]);   // skip row 0 (t-row)
}

// ------------------------- unified WMMA GEMM -------------------------------
// C[128 x 128-col slice] per block: X[m0:m0+128, :KPAD] @ W[KPAD, N].
// 8 waves; wave w owns N-tile (blockIdx.y*8 + w); 8 M-tiles deep in registers.
// K processed in 64-wide LDS panels (128x64 bf16 = 16 KB, shared by 8 waves).
// EPI: 0 = bias+tanh -> bf16 | 1 = bias linear -> f32 | 2 = sech^2 scale -> bf16
//      (EPI==2: rows are 32-per-sample tangent groups; M-blocks are 32-aligned
//       so the scale value is uniform per (block-sample-group, column).)

template <int KPAD, int EPI>
__global__ void __launch_bounds__(256)
k_gemm(const unsigned short* __restrict__ X, const unsigned short* __restrict__ WT,
       const float* __restrict__ bias, const unsigned short* __restrict__ hscale,
       unsigned short* __restrict__ Hout, float* __restrict__ Fout,
       int N, int outStride, int outCols) {
  __shared__ __align__(16) unsigned short As[128 * 64];   // one K-panel
  const int tid  = threadIdx.x;
  const int m0   = blockIdx.x * 128;
  const int wave = tid >> 5, lane = tid & 31;
  const int ntile = blockIdx.y * 8 + wave;
  const bool active = (ntile * 16) < N;                   // wave-uniform

  v8f acc[8];
#pragma unroll
  for (int mt = 0; mt < 8; ++mt) { v8f zz = {}; acc[mt] = zz; }

  constexpr int NPANEL = KPAD / 64;
#pragma unroll 1
  for (int panel = 0; panel < NPANEL; ++panel) {
    // cooperative stage: 128 rows x 64 cols = 1024 x 16B chunks, 4 per thread
#pragma unroll
    for (int it = 0; it < 4; ++it) {
      int ch = tid + it * 256;
      int r = ch >> 3, c = (ch & 7) * 8;
      *(uint4*)(&As[r * 64 + c]) =
          *(const uint4*)(&X[(size_t)(m0 + r) * KPAD + panel * 64 + c]);
    }
    __syncthreads();
    if (active) {
      if (panel + 1 < NPANEL)   // pull next B panel toward the caches early
        __builtin_prefetch(WT + (size_t)(ntile * 16 + (lane & 15)) * KPAD +
                               (panel + 1) * 64, 0, 3);
#pragma unroll
      for (int ks = 0; ks < 2; ++ks) {
        const int k0g = panel * 64 + ks * 32;
        v16bf b = load_b_frag(WT, KPAD, ntile * 16, k0g, lane);
        v16bf a[8];
#pragma unroll
        for (int mt = 0; mt < 8; ++mt)       // batch LDS loads, then compute
          a[mt] = load_a_frag(As + mt * 16 * 64, 64, ks * 32, lane);
#pragma unroll
        for (int mt = 0; mt < 8; ++mt)
          acc[mt] = __builtin_amdgcn_wmma_f32_16x16x32_bf16(false, a[mt], false, b,
                                                            (short)0, acc[mt],
                                                            false, false);
      }
    }
    __syncthreads();
  }

  if (!active) return;
  const int col = lane & 15, hi = lane >> 4;
  const int gcol = ntile * 16 + col;

  if constexpr (EPI == 1) {            // linear, f32 out (heads)
    if (gcol < outCols) {
      float bv = bias[gcol];
#pragma unroll
      for (int mt = 0; mt < 8; ++mt)
#pragma unroll
        for (int g = 0; g < 8; ++g) {
          int row = m0 + mt * 16 + g + 8 * hi;
          Fout[(size_t)row * outStride + gcol] = acc[mt][g] + bv;
        }
    }
  } else if constexpr (EPI == 0) {     // bias + tanh, bf16 out (hidden layers)
    float bv = bias[gcol];
#pragma unroll
    for (int mt = 0; mt < 8; ++mt)
#pragma unroll
      for (int g = 0; g < 8; ++g) {
        int row = m0 + mt * 16 + g + 8 * hi;
        Hout[(size_t)row * N + gcol] = f2bf(tanhf(acc[mt][g] + bv));
      }
  } else {                             // EPI==2: tangent sech^2 scaling
#pragma unroll
    for (int mt = 0; mt < 8; mt += 2) {
      int sample_local = (m0 + mt * 16) >> 5;   // 32 tangent rows per sample
      float hv = bf2f(hscale[(size_t)sample_local * H_N + gcol]);
      float sv = 1.0f - hv * hv;
#pragma unroll
      for (int mm = 0; mm < 2; ++mm)
#pragma unroll
        for (int g = 0; g < 8; ++g) {
          int row = m0 + (mt + mm) * 16 + g + 8 * hi;
          Hout[(size_t)row * H_N + gcol] = f2bf(acc[mt + mm][g] * sv);
        }
    }
  }
}

// ------------------------- tangent init / trace ----------------------------

// T0[(sl*32+k)][j] = W1z[k][j] * (1 - h1[sl][j]^2), one chunk.
__global__ void k_tinit(const unsigned short* __restrict__ W1zb,
                        const unsigned short* __restrict__ h1c,
                        unsigned short* __restrict__ T0) {
  int idx = blockIdx.x * 256 + threadIdx.x;
  if (idx >= CHUNK * 32 * H_N) return;
  int j  = idx & (H_N - 1);
  int rk = idx >> 9;
  int k  = rk & 31;
  int sl = rk >> 5;
  float hv = bf2f(h1c[(size_t)sl * H_N + j]);
  float sv = 1.0f - hv * hv;
  T0[idx] = f2bf(bf2f(W1zb[k * H_N + j]) * sv);
}

// trace[s] = sum_{k<32,j<512} T3[(sl*32+k)][j] * W4T[k][j];  dlogp = g - trace.
__global__ void __launch_bounds__(256)
k_trace(const unsigned short* __restrict__ T3c, const unsigned short* __restrict__ W4T,
        const float* __restrict__ g_in, float* __restrict__ dlogp_out, int chunk0) {
  __shared__ float red[8];
  const int sl = blockIdx.x, tid = threadIdx.x;
  const unsigned short* Ts = T3c + (size_t)sl * 32 * H_N;
  float p = 0.0f;
#pragma unroll 1
  for (int idx = tid * 8; idx < 32 * H_N; idx += 256 * 8) {
    Pk8 a, b;
    a.q = *(const uint4*)(Ts + idx);
    b.q = *(const uint4*)(W4T + idx);
#pragma unroll
    for (int e = 0; e < 8; ++e) p += bf2f(a.s[e]) * bf2f(b.s[e]);
  }
  const int wave = tid >> 5, lane = tid & 31;
#pragma unroll
  for (int off = 16; off > 0; off >>= 1) p += __shfl_down(p, off);
  if (lane == 0) red[wave] = p;
  __syncthreads();
  if (tid == 0) {
    float tr = 0.0f;
#pragma unroll
    for (int w = 0; w < 8; ++w) tr += red[w];
    int s = chunk0 + sl;
    dlogp_out[s] = g_in[s] - tr;
  }
}

// ------------------------- host launcher -----------------------------------

extern "C" void kernel_launch(void* const* d_in, const int* in_sizes, int n_in,
                              void* d_out, int out_size, void* d_ws, size_t ws_size,
                              hipStream_t stream) {
  (void)in_sizes; (void)n_in; (void)out_size; (void)ws_size;

  const float* t = (const float*)d_in[0];
  const float* z = (const float*)d_in[1];
  const float* W[3][4];
  const float* Bv[3][4];
  for (int net = 0; net < 3; ++net)
    for (int l = 0; l < 4; ++l) {
      W[net][l]  = (const float*)d_in[2 + net * 8 + l * 2];
      Bv[net][l] = (const float*)d_in[2 + net * 8 + l * 2 + 1];
    }

  size_t off = 0;
  auto alloc = [&](size_t elems) -> unsigned short* {
    unsigned short* p = (unsigned short*)((char*)d_ws + off);
    off += (elems * sizeof(unsigned short) + 255) & ~(size_t)255;
    return p;
  };

  unsigned short* stateB = alloc((size_t)B_N * 64);
  unsigned short* W1T[3]; unsigned short* W2T[3]; unsigned short* W3T[3]; unsigned short* W4T[3];
  for (int net = 0; net < 3; ++net) {
    W1T[net] = alloc((size_t)H_N * 64);
    W2T[net] = alloc((size_t)H_N * H_N);
    W3T[net] = alloc((size_t)H_N * H_N);
    W4T[net] = alloc((size_t)32 * H_N);          // g-head padded 1 -> 32 cols
  }
  unsigned short* W1zb = alloc((size_t)32 * H_N);
  unsigned short* h1   = alloc((size_t)B_N * H_N);
  unsigned short* h2   = alloc((size_t)B_N * H_N);
  unsigned short* h3   = alloc((size_t)B_N * H_N);
  unsigned short* ping = alloc((size_t)B_N * H_N);
  unsigned short* pong = alloc((size_t)B_N * H_N);
  unsigned short* Ta   = alloc((size_t)CHUNK * 32 * H_N);   // 32 MB
  unsigned short* Tb   = alloc((size_t)CHUNK * 32 * H_N);   // 32 MB

  float* out_dz    = (float*)d_out;                 // [B,32]
  float* out_g     = out_dz + (size_t)B_N * 32;     // [B,1]
  float* out_dlogp = out_g + B_N;                   // [B,1]
  float* out_dlp   = out_dlogp + B_N;               // [B,32]

  // ---- prep ----
  k_build_state<<<(B_N * 64 + 255) / 256, 256, 0, stream>>>(t, z, stateB);
  auto tr = [&](const float* Wp, unsigned short* WTp, int K, int Ns, int KP, int NP) {
    k_transpose<<<(NP * KP + 255) / 256, 256, 0, stream>>>(Wp, WTp, K, Ns, KP, NP);
  };
  for (int net = 0; net < 3; ++net) {
    tr(W[net][0], W1T[net], 33, H_N, 64, H_N);
    tr(W[net][1], W2T[net], H_N, H_N, H_N, H_N);
    tr(W[net][2], W3T[net], H_N, H_N, H_N, H_N);
  }
  tr(W[0][3], W4T[0], H_N, 32, H_N, 32);
  tr(W[1][3], W4T[1], H_N, 1,  H_N, 32);
  tr(W[2][3], W4T[2], H_N, 32, H_N, 32);
  k_cast_w1z<<<(32 * H_N + 255) / 256, 256, 0, stream>>>(W[0][0], W1zb);

  // ---- forward MLPs (M=128 x N=128 blocking, 8 waves/block) ----
  const dim3 blk(256);
  const dim3 gH(B_N / 128, H_N / 128);  // 64 x 4 for N=512
  const dim3 gS(B_N / 128, 1);          // heads (<=32 padded cols)

  // net1 (velocity): keep h1..h3 for tangent pass
  k_gemm<64,  0><<<gH, blk, 0, stream>>>(stateB, W1T[0], Bv[0][0], nullptr, h1, nullptr, H_N, 0, H_N);
  k_gemm<512, 0><<<gH, blk, 0, stream>>>(h1,     W2T[0], Bv[0][1], nullptr, h2, nullptr, H_N, 0, H_N);
  k_gemm<512, 0><<<gH, blk, 0, stream>>>(h2,     W3T[0], Bv[0][2], nullptr, h3, nullptr, H_N, 0, H_N);
  k_gemm<512, 1><<<gS, blk, 0, stream>>>(h3,     W4T[0], Bv[0][3], nullptr, nullptr, out_dz, 32, 32, 32);

  // net2 (g)
  k_gemm<64,  0><<<gH, blk, 0, stream>>>(stateB, W1T[1], Bv[1][0], nullptr, ping, nullptr, H_N, 0, H_N);
  k_gemm<512, 0><<<gH, blk, 0, stream>>>(ping,   W2T[1], Bv[1][1], nullptr, pong, nullptr, H_N, 0, H_N);
  k_gemm<512, 0><<<gH, blk, 0, stream>>>(pong,   W3T[1], Bv[1][2], nullptr, ping, nullptr, H_N, 0, H_N);
  k_gemm<512, 1><<<gS, blk, 0, stream>>>(ping,   W4T[1], Bv[1][3], nullptr, nullptr, out_g, 32, 1, 1);

  // net3 (dlog_p_dz)
  k_gemm<64,  0><<<gH, blk, 0, stream>>>(stateB, W1T[2], Bv[2][0], nullptr, ping, nullptr, H_N, 0, H_N);
  k_gemm<512, 0><<<gH, blk, 0, stream>>>(ping,   W2T[2], Bv[2][1], nullptr, pong, nullptr, H_N, 0, H_N);
  k_gemm<512, 0><<<gH, blk, 0, stream>>>(pong,   W3T[2], Bv[2][2], nullptr, ping, nullptr, H_N, 0, H_N);
  k_gemm<512, 1><<<gS, blk, 0, stream>>>(ping,   W4T[2], Bv[2][3], nullptr, nullptr, out_dlp, 32, 32, 32);

  // ---- exact Jacobian trace, chunked batched tangent GEMMs ----
  const dim3 gT(CHUNK * 32 / 128, H_N / 128);   // 256 x 4
  for (int c = 0; c < NCHUNK; ++c) {
    const int s0 = c * CHUNK;
    k_tinit<<<(CHUNK * 32 * H_N + 255) / 256, 256, 0, stream>>>(
        W1zb, h1 + (size_t)s0 * H_N, Ta);
    k_gemm<512, 2><<<gT, blk, 0, stream>>>(Ta, W2T[0], nullptr,
        h2 + (size_t)s0 * H_N, Tb, nullptr, H_N, 0, H_N);
    k_gemm<512, 2><<<gT, blk, 0, stream>>>(Tb, W3T[0], nullptr,
        h3 + (size_t)s0 * H_N, Ta, nullptr, H_N, 0, H_N);
    k_trace<<<CHUNK, 256, 0, stream>>>(Ta, W4T[0], out_g, out_dlogp, s0);
  }
}
